// KalmanPredictorMambaDirectV1_86062554677328
// MI455X (gfx1250) — matrix operand order repaired and checked by reading, not compiled, not run on hardware
//
#include <hip/hip_runtime.h>
#include <hip/hip_bf16.h>

// ---- problem constants (from reference) ----
#define NLAYER 2
#define Cch    96
#define DIc    192
#define Nst    16
#define Rr     6
#define Bb     4
#define Hh     48
#define Wv     48
#define Ll     (Hh*Wv)     // 2304
#define BLl    (Bb*Ll)     // 9216
#define XROWS  48          // padded rows of x_dbl (38 -> 48)

typedef __attribute__((ext_vector_type(16))) __bf16 v16bf;
typedef __attribute__((ext_vector_type(8)))  float  v8f;

// -----------------------------------------------------------------
// Generic WMMA GEMM: Out[batch] = W[batch%wMod] (MxK) x X[batch] (KxNcols)
// Row-major. One wave computes one 16x16 tile. ALL dims multiples of 16,
// K multiple of 32 -> no bounds checks anywhere (straight-line loads).
// epilogue: 0 = none, 1 = +bias, 2 = +bias then leaky-relu(0.01)
// -----------------------------------------------------------------
__global__ void wmma_gemm_kernel(const float* __restrict__ W, int wMod, long wStride,
                                 const float* __restrict__ X, long xStride,
                                 float* __restrict__ Out, long oStride,
                                 int K, int Ncols,
                                 const float* __restrict__ bias, int epilogue)
{
    const int lane = threadIdx.x & 31;
    const int half = lane >> 4;       // 0: lanes 0-15, 1: lanes 16-31
    const int lm   = lane & 15;
    const int m0   = blockIdx.y * 16;
    const int n0   = blockIdx.x * 16;
    const int bt   = blockIdx.z;

    const float* Wb = W + (long)(bt % wMod) * wStride;
    const float* Xb = X + (long)bt * xStride;
    float*       Ob = Out + (long)bt * oStride;

    const int m = m0 + lm;
    const int n = n0 + lm;
    const float* wrow = Wb + (long)m * K;

    v8f acc = {};
    for (int k0 = 0; k0 < K; k0 += 32) {
        v16bf a, b;
        // A fragment (16x32 bf16): per lane two runs of 8 contiguous floats:
        //   [k0 + half*8 .. +7]  and  [k0 + 16 + half*8 .. +7]   -> 4x float4
        {
            const float4* p0 = (const float4*)(wrow + k0 + half * 8);
            const float4* p1 = (const float4*)(wrow + k0 + 16 + half * 8);
            float4 w0 = p0[0], w1 = p0[1], w2 = p1[0], w3 = p1[1];
            a[0]  = (__bf16)w0.x; a[1]  = (__bf16)w0.y; a[2]  = (__bf16)w0.z; a[3]  = (__bf16)w0.w;
            a[4]  = (__bf16)w1.x; a[5]  = (__bf16)w1.y; a[6]  = (__bf16)w1.z; a[7]  = (__bf16)w1.w;
            a[8]  = (__bf16)w2.x; a[9]  = (__bf16)w2.y; a[10] = (__bf16)w2.z; a[11] = (__bf16)w2.w;
            a[12] = (__bf16)w3.x; a[13] = (__bf16)w3.y; a[14] = (__bf16)w3.z; a[15] = (__bf16)w3.w;
        }
        // B fragment (32x16 bf16): element e -> row k0 + half*8 + (e&7) + (e>>3)*16, col n.
        // Across lanes 0-15 each load covers 16 contiguous columns (coalesced).
        #pragma unroll
        for (int e = 0; e < 16; ++e) {
            b[e] = (__bf16)Xb[(long)(k0 + half * 8 + (e & 7) + (e >> 3) * 16) * Ncols + n];
        }
        acc = __builtin_amdgcn_wmma_f32_16x16x32_bf16(
                  false, a, false, b, (short)0, acc, false, false);
    }
    // C/D layout: VGPR r -> row m0 + r + half*8, col n
    #pragma unroll
    for (int r = 0; r < 8; ++r) {
        int mr = m0 + r + half * 8;
        float v = acc[r];
        if (epilogue) {
            v += bias[mr];
            if (epilogue == 2) v = (v >= 0.0f) ? v : 0.01f * v;
        }
        Ob[(long)mr * Ncols + n] = v;
    }
}

// -----------------------------------------------------------------
// Zero-pad x_proj_w (4,38,192) -> (4,48,192) so the GEMM needs no guards.
// -----------------------------------------------------------------
__global__ void pad_xpw_kernel(const float* __restrict__ src, float* __restrict__ dst)
{
    int idx = blockIdx.x * blockDim.x + threadIdx.x;
    if (idx >= 4 * XROWS * DIc) return;
    int d   = idx % DIc;
    int row = (idx / DIc) % XROWS;
    int k   = idx / (XROWS * DIc);
    dst[idx] = (row < (Rr + 2 * Nst)) ? src[(k * (Rr + 2 * Nst) + row) * DIc + d] : 0.0f;
}

// -----------------------------------------------------------------
// LayerNorm over channel dim (stride Ll) + optional SiLU gate.
// -----------------------------------------------------------------
__global__ void ln_gate_kernel(const float* __restrict__ x, float* __restrict__ out,
                               const float* __restrict__ g, const float* __restrict__ beta,
                               const float* gate, long gbs, int Cn, float eps)
{
    int idx = blockIdx.x * blockDim.x + threadIdx.x;
    if (idx >= BLl) return;
    int b = idx / Ll, l = idx % Ll;
    long base = (long)b * Cn * Ll + l;
    float s = 0.0f, ss = 0.0f;
    for (int c = 0; c < Cn; ++c) {
        float v = x[base + (long)c * Ll];
        s += v; ss += v * v;
    }
    float mu  = s / (float)Cn;
    float var = ss / (float)Cn - mu * mu;
    float inv = rsqrtf(var + eps);
    for (int c = 0; c < Cn; ++c) {
        float v = (x[base + (long)c * Ll] - mu) * inv * g[c] + beta[c];
        if (gate) {
            float zv = gate[(long)b * gbs + (long)c * Ll + l];
            v *= zv / (1.0f + __expf(-zv));
        }
        out[base + (long)c * Ll] = v;
    }
}

// -----------------------------------------------------------------
// Depthwise 3x3 conv (SAME) + bias + SiLU.
// -----------------------------------------------------------------
__global__ void dwconv_silu_kernel(const float* __restrict__ xz, float* __restrict__ xc,
                                   const float* __restrict__ cw, const float* __restrict__ cb)
{
    int idx = blockIdx.x * blockDim.x + threadIdx.x;
    if (idx >= Bb * DIc * Ll) return;
    int l = idx % Ll; int t = idx / Ll; int d = t % DIc; int b = t / DIc;
    int hh = l / Wv, ww = l % Wv;
    const float* src = xz + ((long)b * 2 * DIc + d) * Ll;
    float acc = cb[d];
    #pragma unroll
    for (int dy = -1; dy <= 1; ++dy)
        #pragma unroll
        for (int dx = -1; dx <= 1; ++dx) {
            int y = hh + dy, x2 = ww + dx;
            if (y >= 0 && y < Hh && x2 >= 0 && x2 < Wv)
                acc += cw[d * 9 + (dy + 1) * 3 + (dx + 1)] * src[y * Wv + x2];
        }
    acc = acc / (1.0f + __expf(-acc));   // silu
    xc[((long)b * DIc + d) * Ll + l] = acc;
}

// -----------------------------------------------------------------
// Build 4-direction stack xs (B,4,DIc,Ll) from xc (B,DIc,Ll).
// -----------------------------------------------------------------
__global__ void build_xs_kernel(const float* __restrict__ xc, float* __restrict__ xs)
{
    int idx = blockIdx.x * blockDim.x + threadIdx.x;
    if (idx >= Bb * DIc * Ll) return;
    int l = idx % Ll; int t = idx / Ll; int d = t % DIc; int b = t / DIc;
    float v = xc[idx];
    int hh = l / Wv, ww = l % Wv;
    int p1 = ww * Hh + hh;                 // transposed position
    long bb = (long)b * 4 * DIc * Ll + (long)d * Ll;
    const long ks = (long)DIc * Ll;
    xs[bb + l]                      = v;   // k=0
    xs[bb + ks + p1]                = v;   // k=1
    xs[bb + 2 * ks + (Ll - 1 - l)]  = v;   // k=2
    xs[bb + 3 * ks + (Ll - 1 - p1)] = v;   // k=3
}

// -----------------------------------------------------------------
// delta = softplus( dt_w x dts + dt_b ).  x_dbl has XROWS-row stride.
// -----------------------------------------------------------------
__global__ void delta_kernel(const float* __restrict__ xdbl, float* __restrict__ dly,
                             const float* __restrict__ dw, const float* __restrict__ db)
{
    long idx = (long)blockIdx.x * blockDim.x + threadIdx.x;
    if (idx >= (long)Bb * 4 * DIc * Ll) return;
    int l = (int)(idx % Ll); long t = idx / Ll;
    int d = (int)(t % DIc); t /= DIc;
    int k = (int)(t % 4); int b = (int)(t / 4);
    long xb = (long)(b * 4 + k) * XROWS * Ll;
    float acc = db[k * DIc + d];
    #pragma unroll
    for (int r = 0; r < Rr; ++r)
        acc += dw[(k * DIc + d) * Rr + r] * xdbl[xb + (long)r * Ll + l];
    float sp = (acc > 20.0f) ? acc : __logf(1.0f + __expf(acc));
    dly[idx] = sp;
}

// -----------------------------------------------------------------
// Selective scan. One block per (b,k); lane d keeps h[16] in VGPRs.
// B/C vectors (rows Rr..Rr+2N-1 of x_dbl) staged through LDS each step.
// -----------------------------------------------------------------
__global__ void scan_kernel(const float* __restrict__ xs, const float* __restrict__ xdbl,
                            float* dly,
                            const float* __restrict__ Alog, const float* __restrict__ Ds)
{
    int bk = blockIdx.x;              // b*4+k
    int k  = bk & 3;
    int d  = threadIdx.x;             // 0..191
    float A[Nst], h[Nst];
    #pragma unroll
    for (int n = 0; n < Nst; ++n) {
        A[n] = -__expf(Alog[(k * DIc + d) * Nst + n]);
        h[n] = 0.0f;
    }
    float Dv = Ds[k * DIc + d];
    long base = ((long)bk * DIc + d) * Ll;
    long xb   = (long)bk * XROWS * Ll;
    __shared__ float sB[Nst];
    __shared__ float sC[Nst];
    for (int l = 0; l < Ll; ++l) {
        if (d < Nst)          sB[d]       = xdbl[xb + (long)(Rr + d) * Ll + l];
        else if (d < 2 * Nst) sC[d - Nst] = xdbl[xb + (long)(Rr + d) * Ll + l];
        __syncthreads();
        float dl = dly[base + l];
        float u  = xs[base + l];
        float du = dl * u;
        float y = 0.0f;
        #pragma unroll
        for (int n = 0; n < Nst; ++n) {
            h[n] = h[n] * __expf(dl * A[n]) + du * sB[n];
            y += h[n] * sC[n];
        }
        dly[base + l] = y + Dv * u;
        __syncthreads();
    }
}

// -----------------------------------------------------------------
// Combine 4 directions.
// -----------------------------------------------------------------
__global__ void combine_kernel(const float* __restrict__ oy, float* __restrict__ yg)
{
    int idx = blockIdx.x * blockDim.x + threadIdx.x;
    if (idx >= BLl) return;
    int b = idx / Ll, l = idx % Ll;
    int hh = l / Wv, ww = l % Wv;
    int lt = ww * Hh + hh;                 // transposed source position
    const long ks = (long)DIc * Ll;
    for (int d = 0; d < DIc; ++d) {
        long rb = (long)(b * 4) * ks + (long)d * Ll;
        float v = oy[rb + l]
                + oy[rb + 2 * ks + (Ll - 1 - l)]
                + oy[rb + ks + lt]
                + oy[rb + 3 * ks + (Ll - 1 - lt)];
        yg[((long)b * DIc + d) * Ll + l] = v;
    }
}

// -----------------------------------------------------------------
extern "C" void kernel_launch(void* const* d_in, const int* in_sizes, int n_in,
                              void* d_out, int out_size, void* d_ws, size_t ws_size,
                              hipStream_t stream)
{
    (void)in_sizes; (void)n_in; (void)out_size; (void)ws_size;
    const float* image = (const float*)d_in[0];
    const float* ln1g  = (const float*)d_in[1];
    const float* ln1b  = (const float*)d_in[2];
    const float* inw   = (const float*)d_in[3];
    const float* convw = (const float*)d_in[4];
    const float* convb = (const float*)d_in[5];
    const float* xpw   = (const float*)d_in[6];
    const float* dtw   = (const float*)d_in[7];
    const float* dtb   = (const float*)d_in[8];
    const float* Alog  = (const float*)d_in[9];
    const float* Dsp   = (const float*)d_in[10];
    const float* ong   = (const float*)d_in[11];
    const float* onb   = (const float*)d_in[12];
    const float* opw   = (const float*)d_in[13];
    const float* ln2g  = (const float*)d_in[14];
    const float* ln2b  = (const float*)d_in[15];
    const float* m1w   = (const float*)d_in[16];
    const float* m1b   = (const float*)d_in[17];
    const float* m2w   = (const float*)d_in[18];
    const float* m2b   = (const float*)d_in[19];

    float* ws    = (float*)d_ws;
    float* xln   = ws;                          // B*96*L
    float* xz    = xln  + (long)Bb*Cch*Ll;      // B*384*L
    float* xc    = xz   + (long)Bb*2*DIc*Ll;    // B*192*L
    float* xs    = xc   + (long)Bb*DIc*Ll;      // B*4*192*L
    float* xpwp  = xs   + (long)Bb*4*DIc*Ll;    // 4*48*192 padded weights
    float* xdbl  = xpwp + (long)4*XROWS*DIc;    // B*4*48*L
    float* dly   = xdbl + (long)Bb*4*XROWS*Ll;  // B*4*192*L (delta -> out_y in place)
    float* yg    = dly  + (long)Bb*4*DIc*Ll;    // B*192*L
    float* xcur  = yg   + (long)Bb*DIc*Ll;      // B*96*L
    float* xtmp  = xcur + (long)Bb*Cch*Ll;      // B*96*L

    const int NT = Ll / 16;                     // 144 column tiles
    const float* src = image;

    for (int i = 0; i < NLAYER; ++i) {
        // 1) LN1 over C=96
        ln_gate_kernel<<<(BLl + 127) / 128, 128, 0, stream>>>(
            src, xln, ln1g + i * Cch, ln1b + i * Cch, nullptr, 0, Cch, 1e-6f);

        // 2) in_proj: xz(B,384,L) = W(384x96) x xln(96,L)   [WMMA]
        {
            dim3 g(NT, 2 * DIc / 16, Bb);
            wmma_gemm_kernel<<<g, 32, 0, stream>>>(
                inw + (long)i * 2 * DIc * Cch, 1, 0,
                xln, (long)Cch * Ll,
                xz, (long)2 * DIc * Ll,
                Cch, Ll, nullptr, 0);
        }

        // 3) depthwise 3x3 conv + bias + silu
        dwconv_silu_kernel<<<(Bb * DIc * Ll + 255) / 256, 256, 0, stream>>>(
            xz, xc, convw + (long)i * DIc * 9, convb + i * DIc);

        // 4) 4-direction stack
        build_xs_kernel<<<(Bb * DIc * Ll + 255) / 256, 256, 0, stream>>>(xc, xs);

        // 5) pad x_proj_w rows 38 -> 48 (removes all guards from the GEMM)
        pad_xpw_kernel<<<(4 * XROWS * DIc + 255) / 256, 256, 0, stream>>>(
            xpw + (long)i * 4 * (Rr + 2 * Nst) * DIc, xpwp);

        // 6) x_dbl: per (b,k)  (48,L) = xw_pad[k](48x192) x xs[b,k](192,L)  [WMMA]
        {
            dim3 g(NT, XROWS / 16, Bb * 4);
            wmma_gemm_kernel<<<g, 32, 0, stream>>>(
                xpwp, 4, (long)XROWS * DIc,
                xs, (long)DIc * Ll,
                xdbl, (long)XROWS * Ll,
                DIc, Ll, nullptr, 0);
        }

        // 7) delta = softplus(dt_w x dts + dt_b)
        {
            long tot = (long)Bb * 4 * DIc * Ll;
            delta_kernel<<<(unsigned)((tot + 255) / 256), 256, 0, stream>>>(
                xdbl, dly, dtw + (long)i * 4 * DIc * Rr, dtb + (long)i * 4 * DIc);
        }

        // 8) selective scan (in-place delta -> out_y)
        scan_kernel<<<Bb * 4, DIc, 0, stream>>>(
            xs, xdbl, dly, Alog + (long)i * 4 * DIc * Nst, Dsp + (long)i * 4 * DIc);

        // 9) combine 4 directions
        combine_kernel<<<(BLl + 127) / 128, 128, 0, stream>>>(dly, yg);

        // 10) out-norm (eps 1e-5) + SiLU(z) gate, in place on yg
        ln_gate_kernel<<<(BLl + 127) / 128, 128, 0, stream>>>(
            yg, yg, ong + i * DIc, onb + i * DIc,
            xz + (long)DIc * Ll, (long)2 * DIc * Ll, DIc, 1e-5f);

        // 11) out_proj: xcur(B,96,L) = ow(96x192) x yg(192,L)  [WMMA]
        {
            dim3 g(NT, Cch / 16, Bb);
            wmma_gemm_kernel<<<g, 32, 0, stream>>>(
                opw + (long)i * Cch * DIc, 1, 0,
                yg, (long)DIc * Ll,
                xcur, (long)Cch * Ll,
                DIc, Ll, nullptr, 0);
        }

        // 12) LN2
        ln_gate_kernel<<<(BLl + 127) / 128, 128, 0, stream>>>(
            xcur, xln, ln2g + i * Cch, ln2b + i * Cch, nullptr, 0, Cch, 1e-6f);

        // 13) MLP1 + bias + leaky-relu  [WMMA, fused epilogue]
        {
            dim3 g(NT, Cch / 16, Bb);
            wmma_gemm_kernel<<<g, 32, 0, stream>>>(
                m1w + (long)i * Cch * Cch, 1, 0,
                xln, (long)Cch * Ll,
                xtmp, (long)Cch * Ll,
                Cch, Ll, m1b + i * Cch, 2);
        }

        // 14) MLP2 + bias  [WMMA]; final layer writes d_out directly
        {
            float* dst = (i == NLAYER - 1) ? (float*)d_out : xcur;
            dim3 g(NT, Cch / 16, Bb);
            wmma_gemm_kernel<<<g, 32, 0, stream>>>(
                m2w + (long)i * Cch * Cch, 1, 0,
                xtmp, (long)Cch * Ll,
                dst, (long)Cch * Ll,
                Cch, Ll, m2b + i * Cch, 1);
        }

        src = xcur;
    }
}
